// MoEFFN_82772609728618
// MI455X (gfx1250) — matrix operand order repaired and checked by reading, not compiled, not run on hardware
//
#include <hip/hip_runtime.h>
#include <hip/hip_bf16.h>
#include <math.h>

// ---------------- problem constants (from reference) ----------------
#define T_TOK 16384          // B*S = 8*2048
#define H_DIM 768
#define I_DIM 3072
#define E_EXP 8
#define TILE_M 32            // tokens per workgroup tile
#define CHUNK 64             // I-dim chunk per fused iteration
#define SX 784               // LDS row stride for x tile (pad: 768+16, 16B aligned, breaks bank conflicts)
#define SH 72                // LDS row stride for h chunk (64+8, 16B aligned)

typedef __attribute__((ext_vector_type(16))) _Float16 v16h;
typedef __attribute__((ext_vector_type(8)))  float    v8f;

// two 16-byte chunks -> one 16-element f16 fragment
static __device__ __forceinline__ v16h load_frag16(const _Float16* p0, const _Float16* p1) {
    union { v16h v; uint4 q[2]; } u;
    u.q[0] = *(const uint4*)p0;
    u.q[1] = *(const uint4*)p1;
    return u.v;
}

static __device__ __forceinline__ float gelu_exact(float x) {
    return 0.5f * x * (1.0f + erff(x * 0.70710678118654752440f));
}

// ---------------- kernel 1: zero output + counters ----------------
__global__ void zero_kernel(float* __restrict__ out, int n, int* __restrict__ counts) {
    int i = blockIdx.x * blockDim.x + threadIdx.x;
    for (int k = i; k < n; k += gridDim.x * blockDim.x) out[k] = 0.0f;
    if (i < E_EXP) counts[i] = 0;
}

// ---------------- kernel 2: x f32 -> f16 ----------------
__global__ void cvt_x_kernel(const float* __restrict__ x, _Float16* __restrict__ xb, int n) {
    int i = blockIdx.x * blockDim.x + threadIdx.x;
    for (int k = i; k < n; k += gridDim.x * blockDim.x) xb[k] = (_Float16)x[k];
}

// ---------------- kernel 3: up_w [E][H][I] f32 -> upT [E][I][H] f16 ----------------
__global__ void cvt_up_kernel(const float* __restrict__ up, _Float16* __restrict__ upT) {
    const long n = (long)E_EXP * I_DIM * H_DIM;
    long i = (long)blockIdx.x * blockDim.x + threadIdx.x;
    const long stride = (long)gridDim.x * blockDim.x;
    for (long o = i; o < n; o += stride) {
        long e   = o / ((long)I_DIM * H_DIM);
        long rem = o - e * (long)I_DIM * H_DIM;
        long ii  = rem / H_DIM;
        long h   = rem - ii * H_DIM;
        upT[o] = (_Float16)up[(e * H_DIM + h) * I_DIM + ii];
    }
}

// ---------------- kernel 4: down_w [E][I][H] f32 -> dnT [E][H][I] f16 ----------------
__global__ void cvt_dn_kernel(const float* __restrict__ dn, _Float16* __restrict__ dnT) {
    const long n = (long)E_EXP * H_DIM * I_DIM;
    long i = (long)blockIdx.x * blockDim.x + threadIdx.x;
    const long stride = (long)gridDim.x * blockDim.x;
    for (long o = i; o < n; o += stride) {
        long e   = o / ((long)H_DIM * I_DIM);
        long rem = o - e * (long)H_DIM * I_DIM;
        long h   = rem / I_DIM;
        long ii  = rem - h * I_DIM;
        dnT[o] = (_Float16)dn[(e * I_DIM + ii) * H_DIM + h];
    }
}

// ---------------- kernel 5: router (softmax + top2 + renorm + bucketing) ----------------
__global__ __launch_bounds__(128) void router_kernel(
        const float* __restrict__ x, const float* __restrict__ rw,
        float* __restrict__ gate, int* __restrict__ counts, int* __restrict__ lists) {
    const int token = blockIdx.x * 4 + (threadIdx.x >> 5);
    const int lane  = threadIdx.x & 31;
    const float* xr = x + (size_t)token * H_DIM;

    float s[E_EXP];
    #pragma unroll
    for (int e = 0; e < E_EXP; ++e) s[e] = 0.0f;
    for (int h = lane; h < H_DIM; h += 32) {
        float xv = xr[h];
        #pragma unroll
        for (int e = 0; e < E_EXP; ++e) s[e] += xv * rw[e * H_DIM + h];
    }
    #pragma unroll
    for (int e = 0; e < E_EXP; ++e) {
        #pragma unroll
        for (int off = 16; off > 0; off >>= 1) s[e] += __shfl_down(s[e], off, 32);
    }
    if (lane == 0) {
        float m = s[0];
        #pragma unroll
        for (int e = 1; e < E_EXP; ++e) m = fmaxf(m, s[e]);
        float p[E_EXP];
        #pragma unroll
        for (int e = 0; e < E_EXP; ++e) p[e] = __expf(s[e] - m);
        // top-2 (ties -> lower index, like jax top_k)
        int i1 = 0;
        #pragma unroll
        for (int e = 1; e < E_EXP; ++e) if (p[e] > p[i1]) i1 = e;
        int i2 = (i1 == 0) ? 1 : 0;
        #pragma unroll
        for (int e = 0; e < E_EXP; ++e) if (e != i1 && p[e] > p[i2]) i2 = e;
        float inv = 1.0f / (p[i1] + p[i2]);     // softmax denom cancels in renorm
        #pragma unroll
        for (int e = 0; e < E_EXP; ++e) gate[token * E_EXP + e] = 0.0f;
        gate[token * E_EXP + i1] = p[i1] * inv;
        gate[token * E_EXP + i2] = p[i2] * inv;
        int q1 = atomicAdd(&counts[i1], 1); lists[i1 * T_TOK + q1] = token;
        int q2 = atomicAdd(&counts[i2], 1); lists[i2 * T_TOK + q2] = token;
    }
}

// ---------------- kernel 6: fused expert FFN via WMMA ----------------
// grid = (512 tiles, E experts), block = 256 (8 waves)
__global__ __launch_bounds__(256) void moe_kernel(
        const _Float16* __restrict__ xb,  const _Float16* __restrict__ upT,
        const _Float16* __restrict__ dnT, const float* __restrict__ gate,
        const int* __restrict__ counts,   const int* __restrict__ lists,
        float* __restrict__ out) {
    const int e   = blockIdx.y;
    const int cnt = counts[e];
    const int t0  = blockIdx.x * TILE_M;
    if (t0 >= cnt) return;

    __shared__ __align__(16) _Float16 xs[TILE_M * SX];   // x tile, f16, padded
    __shared__ __align__(16) _Float16 hs[TILE_M * SH];   // gelu(x@up) chunk, f16
    __shared__ int   ts[TILE_M];
    __shared__ float gs[TILE_M];

    const int tid = threadIdx.x;
    if (tid < TILE_M) {
        int idx = t0 + tid;
        int t   = (idx < cnt) ? lists[e * T_TOK + idx] : -1;
        ts[tid] = t;
        gs[tid] = (t >= 0) ? gate[t * E_EXP + e] : 0.0f;
    }
    __syncthreads();

    // gather x tile into LDS (8 threads per row, 12 x uint4 each)
    {
        int row  = tid >> 3;
        int part = tid & 7;
        int t    = ts[row];
        const uint4* src = (t >= 0) ? (const uint4*)(xb + (size_t)t * H_DIM) : nullptr;
        uint4 z; z.x = z.y = z.z = z.w = 0u;
        #pragma unroll
        for (int j = 0; j < 12; ++j) {
            int q = part * 12 + j;                          // 16B chunk index in row
            uint4 v = src ? src[q] : z;
            *(uint4*)&xs[row * SX + q * 8] = v;
        }
    }
    __syncthreads();

    const int wave = tid >> 5;
    const int lane = tid & 31;
    const int half = lane >> 4;
    const int l15  = lane & 15;
    const int tmA  = wave >> 2, nwA = wave & 3;   // stage-A: 2x4 tiles of 16x16 over [32 x 64]
    const int tmB  = wave >> 2, ngB = wave & 3;   // stage-B: wave owns 12 n-tiles (192 cols)

    const _Float16* upe = upT + (size_t)e * I_DIM * H_DIM;  // [I][H]
    const _Float16* dne = dnT + (size_t)e * H_DIM * I_DIM;  // [H][I]

    v8f acc[12];
    #pragma unroll
    for (int j = 0; j < 12; ++j)
        #pragma unroll
        for (int r = 0; r < 8; ++r) acc[j][r] = 0.0f;

    const _Float16* arowX = &xs[(tmA * 16 + l15) * SX];
    const _Float16* arowH = &hs[(tmB * 16 + l15) * SH];

    for (int ic = 0; ic < I_DIM / CHUNK; ++ic) {
        // ---- stage A: d = x_tile(16xH) @ upT rows (16 cols), K = 768 ----
        v8f d = {};
        {
            const int n = ic * CHUNK + nwA * 16 + l15;
            const _Float16* brow = upe + (size_t)n * H_DIM;
            for (int kk = 0; kk < H_DIM; kk += 32) {
                v16h a = load_frag16(arowX + kk + half * 8, arowX + kk + 16 + half * 8);
                const _Float16* bp = brow + kk + half * 8;
                v16h b = load_frag16(bp, bp + 16);
                d = __builtin_amdgcn_wmma_f32_16x16x32_f16(false, a, false, b,
                                                           (short)0, d, false, false);
            }
        }
        __syncthreads();   // previous iteration's hs consumers are done
        // gelu + D-layout -> row-major f16 in LDS
        #pragma unroll
        for (int r = 0; r < 8; ++r) {
            int mrow = tmA * 16 + half * 8 + r;
            hs[mrow * SH + nwA * 16 + l15] = (_Float16)gelu_exact(d[r]);
        }
        __syncthreads();   // hs ready

        // ---- stage B: acc += h_chunk(16x64) @ dnT rows, 12 n-tiles, K = 64 ----
        v16h a0 = load_frag16(arowH + 0  + half * 8, arowH + 16 + half * 8);
        v16h a1 = load_frag16(arowH + 32 + half * 8, arowH + 48 + half * 8);
        #pragma unroll
        for (int j = 0; j < 12; ++j) {
            int n = ngB * 192 + j * 16 + l15;
            const _Float16* br = dne + (size_t)n * I_DIM + ic * CHUNK + half * 8;
            v16h b0 = load_frag16(br,      br + 16);
            v16h b1 = load_frag16(br + 32, br + 48);
            acc[j] = __builtin_amdgcn_wmma_f32_16x16x32_f16(false, a0, false, b0,
                                                            (short)0, acc[j], false, false);
            acc[j] = __builtin_amdgcn_wmma_f32_16x16x32_f16(false, a1, false, b1,
                                                            (short)0, acc[j], false, false);
        }
    }

    // ---- epilogue: scale rows by gate, atomic-accumulate into out ----
    #pragma unroll
    for (int j = 0; j < 12; ++j) {
        int n = ngB * 192 + j * 16 + l15;
        #pragma unroll
        for (int r = 0; r < 8; ++r) {
            int m = tmB * 16 + half * 8 + r;
            int t = ts[m];
            if (t >= 0) atomicAdd(&out[(size_t)t * H_DIM + n], acc[j][r] * gs[m]);
        }
    }
}

// ---------------- host-side launcher ----------------
extern "C" void kernel_launch(void* const* d_in, const int* in_sizes, int n_in,
                              void* d_out, int out_size, void* d_ws, size_t ws_size,
                              hipStream_t stream) {
    const float* x   = (const float*)d_in[0];   // [T, H]
    const float* rw  = (const float*)d_in[1];   // [E, H]
    const float* up  = (const float*)d_in[2];   // [E, H, I]
    const float* dn  = (const float*)d_in[3];   // [E, I, H]
    float* out = (float*)d_out;                 // [T, H]

    // workspace layout (bytes)
    char* ws = (char*)d_ws;
    const size_t sz_xb  = (size_t)T_TOK * H_DIM * 2;                 // 25.2 MB
    const size_t sz_w   = (size_t)E_EXP * I_DIM * H_DIM * 2;         // 37.7 MB each
    const size_t sz_g   = (size_t)T_TOK * E_EXP * 4;                 // 0.5 MB
    const size_t sz_cnt = 256;
    _Float16* xb    = (_Float16*)(ws);
    _Float16* upT   = (_Float16*)(ws + sz_xb);
    _Float16* dnT   = (_Float16*)(ws + sz_xb + sz_w);
    float*    gate  = (float*)   (ws + sz_xb + 2 * sz_w);
    int*      cnts  = (int*)     (ws + sz_xb + 2 * sz_w + sz_g);
    int*      lists = (int*)     (ws + sz_xb + 2 * sz_w + sz_g + sz_cnt);

    zero_kernel  <<<2048, 256, 0, stream>>>(out, T_TOK * H_DIM, cnts);
    cvt_x_kernel <<<4096, 256, 0, stream>>>(x, xb, T_TOK * H_DIM);
    cvt_up_kernel<<<8192, 256, 0, stream>>>(up, upT);
    cvt_dn_kernel<<<8192, 256, 0, stream>>>(dn, dnT);
    router_kernel<<<T_TOK / 4, 128, 0, stream>>>(x, rw, gate, cnts, lists);
    moe_kernel   <<<dim3(T_TOK / TILE_M, E_EXP), 256, 0, stream>>>(
        xb, upT, dnT, gate, cnts, lists, out);
}
